// CRF_64192581206401
// MI455X (gfx1250) — matrix operand order repaired and checked by reading, not compiled, not run on hardware
//
#include <hip/hip_runtime.h>

typedef __attribute__((ext_vector_type(16))) _Float16     v16h;
typedef __attribute__((ext_vector_type(2)))  _Float16     h2;
typedef __attribute__((ext_vector_type(8)))  float        v8f;
typedef __attribute__((ext_vector_type(8)))  unsigned int v8u;
typedef __attribute__((ext_vector_type(4)))  float        f4;

#define T_LEN   2048
#define C_DIM   64
#define PF_DIST 8

#if __has_builtin(__builtin_amdgcn_exp2f)
#define EXP2F(x) __builtin_amdgcn_exp2f(x)
#else
#define EXP2F(x) exp2f(x)
#endif
#if __has_builtin(__builtin_amdgcn_logf)
#define LOG2F(x) __builtin_amdgcn_logf(x)
#else
#define LOG2F(x) log2f(x)
#endif

// lane i <-> lane i^16 full-width swap, kept on the VALU (v_permlanex16_b32)
__device__ __forceinline__ float xswap16(float x) {
#if __has_builtin(__builtin_amdgcn_permlanex16)
    int r = __builtin_amdgcn_permlanex16(
        __builtin_bit_cast(int, x), __builtin_bit_cast(int, x),
        0x76543210, 0xfedcba98, false, false);
    return __builtin_bit_cast(float, r);
#else
    return __shfl_xor(x, 16);
#endif
}

#define LOG2E 1.4426950408889634f
#define LN2   0.6931471805599453f

// Transposed CRF scan: one wave owns 16 batches; lane (l, half hh) holds
// batch b = b0 + l.  State beta^T in WMMA C/D layout of D = S^T with the
// permuted class map  class(jt, r, hh) = (jt&1)*32 + hh*16 + (jt>>1)*8 + r,
// chosen so each half-wave owns exactly the K-ranges the B operand needs:
// B fragments are assembled by pure in-register f16 packing (no shuffles).
__launch_bounds__(32)
__global__ void crf_fwd_wmma(const float* __restrict__ pad_x,
                             const float* __restrict__ trans,
                             const float* __restrict__ orig,
                             const int*   __restrict__ bsz,
                             float*       __restrict__ out)
{
    const int lane = threadIdx.x;          // 0..31 (wave32)
    const int l    = lane & 15;
    const int hh   = lane >> 4;            // class-half of the storage map
    const int b    = blockIdx.x * 16 + l;  // batch owned by this lane

    // ---- per-lane constants -------------------------------------------------
    const int bs = bsz[b];
    int tmax = bs;
    tmax = max(tmax, __shfl_xor(tmax, 1));
    tmax = max(tmax, __shfl_xor(tmax, 2));
    tmax = max(tmax, __shfl_xor(tmax, 4));
    tmax = max(tmax, __shfl_xor(tmax, 8));
    tmax = max(tmax, __shfl_xor(tmax, 16));
    tmax = __builtin_amdgcn_readfirstlane(tmax);

    // ---- A operand fragments (persistent): A[m,K] = exp(trans[j(m,jt), K]) --
    // 16-bit A 16x32, contiguous k order used here: row m = lane%16; lanes 0-15 K={0..15},
    // lanes 16-31 K={16..31} (+kt*32).  k-index map is the identity;
    // row m of tile jt is class (jt&1)*32 + (m>>3)*16 + (jt>>1)*8 + (m&7).
    v16h afrag[2][4];
    #pragma unroll
    for (int kt = 0; kt < 2; ++kt)
      #pragma unroll
      for (int jt = 0; jt < 4; ++jt) {
          const int j = (jt & 1) * 32 + ((l >> 3) & 1) * 16
                      + (jt >> 1) * 8 + (l & 7);
          #pragma unroll
          for (int v = 0; v < 8; ++v)
            #pragma unroll
            for (int e = 0; e < 2; ++e) {
                int K = kt * 32 + hh * 16 + v * 2 + e;
                afrag[kt][jt][2 * v + e] =
                    (_Float16)EXP2F(trans[j * C_DIM + K] * LOG2E);
            }
      }

    // per-lane batch base; x/orig slice for (jt,hh) is 8 contiguous floats
    const float* xb = pad_x + (size_t)b * T_LEN * C_DIM;
    int cbase[4];
    #pragma unroll
    for (int jt = 0; jt < 4; ++jt)
        cbase[jt] = (jt & 1) * 32 + hh * 16 + (jt >> 1) * 8;
    // prefetch map: 16 batches x 2 cachelines per t, one line per lane
    const char* pfbase = (const char*)xb + (size_t)hh * 128;

    // ---- t = 0: beta = (x0 + origination) * log2e ---------------------------
    v8f beta[4];
    #pragma unroll
    for (int jt = 0; jt < 4; ++jt) {
        const f4* xp = (const f4*)(xb + cbase[jt]);
        f4 x0 = xp[0], x1 = xp[1];
        #pragma unroll
        for (int r = 0; r < 8; ++r) {
            float xvv = (r < 4) ? x0[r] : x1[r - 4];
            beta[jt][r] = (xvv + orig[cbase[jt] + r]) * LOG2E;
        }
    }

    // per-lane partial (own 32 classes) latched at t == bs-1
    float pfin;
    {
        float s[4];
        #pragma unroll
        for (int jt = 0; jt < 4; ++jt) {
            float a0 = beta[jt][0] + beta[jt][1], a1 = beta[jt][2] + beta[jt][3];
            float a2 = beta[jt][4] + beta[jt][5], a3 = beta[jt][6] + beta[jt][7];
            s[jt] = (a0 + a1) + (a2 + a3);
        }
        float p = (s[0] + s[1]) + (s[2] + s[3]);
        pfin = (bs == 1) ? p : 0.0f;
    }

    // ---- sequential scan over T ---------------------------------------------
    for (int t = 1; t < tmax; ++t) {
        // x_t: 8 contiguous floats per jt -> 2 x b128 loads, consumed late
        f4 xv[4][2];
        #pragma unroll
        for (int jt = 0; jt < 4; ++jt) {
            const f4* xp = (const f4*)(xb + (size_t)t * C_DIM + cbase[jt]);
            xv[jt][0] = xp[0];
            xv[jt][1] = xp[1];
        }
        if (t + PF_DIST < T_LEN)
            __builtin_prefetch(pfbase + (size_t)(t + PF_DIST) * C_DIM * 4, 0, 0);

        // per-batch rescale: max over own 32 classes + half-lane partner's 32
        float mjt[4];
        #pragma unroll
        for (int jt = 0; jt < 4; ++jt) {
            float a0 = fmaxf(beta[jt][0], beta[jt][1]);
            float a1 = fmaxf(beta[jt][2], beta[jt][3]);
            float a2 = fmaxf(beta[jt][4], beta[jt][5]);
            float a3 = fmaxf(beta[jt][6], beta[jt][7]);
            mjt[jt] = fmaxf(fmaxf(a0, a1), fmaxf(a2, a3));
        }
        float mv = fmaxf(fmaxf(mjt[0], mjt[1]), fmaxf(mjt[2], mjt[3]));
        mv = fmaxf(mv, xswap16(mv));   // bit-identical on both half-lanes

        // B operand fragments: pure in-register packing.  Fragment kt:
        // VGPR v   = f16{e[kt][2v],   e[kt][2v+1]}
        // VGPR v+4 = f16{e[kt+2][2v], e[kt+2][2v+1]}
        float e[4][8];
        #pragma unroll
        for (int jt = 0; jt < 4; ++jt)
          #pragma unroll
          for (int r = 0; r < 8; ++r)
              e[jt][r] = EXP2F(beta[jt][r] - mv);

        v16h bfrag[2];
        #pragma unroll
        for (int kt = 0; kt < 2; ++kt) {
            v8u bw;
            #pragma unroll
            for (int v = 0; v < 4; ++v) {
                h2 plo, phi;
                plo.x = (_Float16)e[kt][2 * v];
                plo.y = (_Float16)e[kt][2 * v + 1];
                phi.x = (_Float16)e[kt + 2][2 * v];
                phi.y = (_Float16)e[kt + 2][2 * v + 1];
                bw[v]     = __builtin_bit_cast(unsigned int, plo);
                bw[v + 4] = __builtin_bit_cast(unsigned int, phi);
            }
            bfrag[kt] = __builtin_bit_cast(v16h, bw);
        }

        // S^T = exp(trans) x E^T  (K=64 via two chained K=32 WMMAs), then
        // beta' = fma(x, log2e, m) + log2(S)   (raw v_log_f32: S >= 1, safe)
        #pragma unroll
        for (int jt = 0; jt < 4; ++jt) {
            v8f acc = {};
            acc = __builtin_amdgcn_wmma_f32_16x16x32_f16(
                false, afrag[0][jt], false, bfrag[0], (short)0, acc, false, false);
            acc = __builtin_amdgcn_wmma_f32_16x16x32_f16(
                false, afrag[1][jt], false, bfrag[1], (short)0, acc, false, false);
            #pragma unroll
            for (int r = 0; r < 8; ++r) {
                float xvv = (r < 4) ? xv[jt][0][r] : xv[jt][1][r - 4];
                beta[jt][r] = fmaf(xvv, LOG2E, mv) + LOG2F(acc[r]);
            }
        }

        // latch per-lane partial sum at t == bs-1 (branchless)
        {
            float s[4];
            #pragma unroll
            for (int jt = 0; jt < 4; ++jt) {
                float a0 = beta[jt][0] + beta[jt][1], a1 = beta[jt][2] + beta[jt][3];
                float a2 = beta[jt][4] + beta[jt][5], a3 = beta[jt][6] + beta[jt][7];
                s[jt] = (a0 + a1) + (a2 + a3);
            }
            float p = (s[0] + s[1]) + (s[2] + s[3]);
            pfin = (bs == t + 1) ? p : pfin;
        }
    }

    // combine the two half-lane partials of each batch; back to natural log
    pfin += xswap16(pfin);
    if (hh == 0)
        out[b] = pfin * LN2;
}

extern "C" void kernel_launch(void* const* d_in, const int* in_sizes, int n_in,
                              void* d_out, int out_size, void* d_ws, size_t ws_size,
                              hipStream_t stream)
{
    const float* pad_x = (const float*)d_in[0];   // (256, 2048, 64) f32
    const float* trans = (const float*)d_in[1];   // (64, 64) f32
    const float* orig  = (const float*)d_in[2];   // (64,) f32
    const int*   bsz   = (const int*)d_in[3];     // (256,) i32
    float* out = (float*)d_out;                   // (256,) f32

    crf_fwd_wmma<<<dim3(16), dim3(32), 0, stream>>>(pad_x, trans, orig, bsz, out);
}
